// SPY_84911503442759
// MI455X (gfx1250) — matrix operand order repaired and checked by reading, compiled, not run on hardware
//
#include <hip/hip_runtime.h>

#define B_    16
#define N_    512
#define H_    128
#define E_    4
#define STEPS_ 5
#define NE_   (N_*E_)     // 2048
#define ROWS_ (B_*N_)     // 8192
#define K3H_  (3*H_)      // 384
#define KF_   160         // 129 padded to multiple of 32

typedef __attribute__((ext_vector_type(16))) __bf16 v16bf;
typedef __attribute__((ext_vector_type(8)))  __bf16 v8bf;
typedef __attribute__((ext_vector_type(8)))  float  v8f;
typedef __attribute__((ext_vector_type(4)))  float  v4f;

static __device__ __forceinline__ v16bf cat8(v8bf lo, v8bf hi) {
  return __builtin_shufflevector(lo, hi, 0,1,2,3,4,5,6,7,8,9,10,11,12,13,14,15);
}

// ---- WMMA fragment loaders (wave32, 16x16x32 bf16) ------------------------
// B fragment: B stored transposed bt[Ncols][stride]; lane L col = n0+L%16,
// elements 0..7 -> K = k0+(L/16)*8..+7, elements 8..15 -> +16.
static __device__ __forceinline__ v16bf load_bfrag(const __bf16* bt, int stride,
                                                   int n0, int k0, int lane) {
  const __bf16* p = bt + (size_t)(n0 + (lane & 15)) * stride + k0 + ((lane >> 4) << 3);
  return cat8(*(const v8bf*)p, *(const v8bf*)(p + 16));
}

// A fragment from bf16 row-major a[M][stride]: lane L row = m0+L%16,
// elements 0..7 -> K = k0+(L/16)*8..+7, elements 8..15 -> +16.
static __device__ __forceinline__ v16bf load_afrag(const __bf16* a, int stride,
                                                   int m0, int k0, int lane) {
  const __bf16* p = a + (size_t)(m0 + (lane & 15)) * stride + k0 + ((lane >> 4) << 3);
  return cat8(*(const v8bf*)p, *(const v8bf*)(p + 16));
}

// A fragment converted on the fly from f32 row-major (for adjacency matrix m)
static __device__ __forceinline__ v16bf load_afrag_f32(const float* a, int stride,
                                                       int m0, int k0, int lane) {
  const float* p = a + (size_t)(m0 + (lane & 15)) * stride + k0 + ((lane >> 4) << 3);
  v4f x0 = *(const v4f*)p, x1 = *(const v4f*)(p + 4);
  v4f x2 = *(const v4f*)(p + 16), x3 = *(const v4f*)(p + 20);
  v16bf r;
#pragma unroll
  for (int i = 0; i < 4; ++i) {
    r[i]      = (__bf16)x0[i];
    r[4 + i]  = (__bf16)x1[i];
    r[8 + i]  = (__bf16)x2[i];
    r[12 + i] = (__bf16)x3[i];
  }
  return r;
}

static __device__ __forceinline__ v8f wmma_bf16(v16bf a, v16bf b, v8f c) {
  return __builtin_amdgcn_wmma_f32_16x16x32_bf16(false, a, false, b, (short)0, c,
                                                 false, false);
}

static __device__ __forceinline__ float sigmoidf_(float x) {
  return 1.0f / (1.0f + __expf(-x));
}

// ---- elementwise helpers ---------------------------------------------------
__global__ void k_copy_f32(float* dst, const float* src, int n) {
  int i = blockIdx.x * blockDim.x + threadIdx.x;
  if (i < n) dst[i] = src[i];
}

__global__ void k_f32_to_bf(__bf16* dst, const float* src, int n) {
  int i = blockIdx.x * blockDim.x + threadIdx.x;
  if (i < n) dst[i] = (__bf16)src[i];
}

// dst[n][k] = (k < K) ? src[k][n] : 0 ; src shape (K, Ncols), dst (Ncols, Kpad)
__global__ void k_transpose_bf(__bf16* dst, const float* src, int K, int Ncols, int Kpad) {
  int i = blockIdx.x * blockDim.x + threadIdx.x;
  if (i >= Ncols * Kpad) return;
  int n = i / Kpad, k = i - n * Kpad;
  dst[i] = (k < K) ? (__bf16)src[(size_t)k * Ncols + n] : (__bf16)0.0f;
}

// ---- GEMM 1: states = h @ W_{in,out} + b, scattered+transposed to
// statesT[b][h'][e*N + n]  (== _edge_states, transposed for B-frag loads)
__global__ void k_states_gemm(const __bf16* __restrict__ hbf,
                              const __bf16* __restrict__ WinT,
                              const __bf16* __restrict__ WoutT,
                              const float* __restrict__ b_in,
                              const float* __restrict__ b_out,
                              __bf16* __restrict__ inT,
                              __bf16* __restrict__ outT) {
  int sel  = blockIdx.z;
  const __bf16* wt   = sel ? WoutT : WinT;
  const float*  bias = sel ? b_out : b_in;
  __bf16*       dstT = sel ? outT  : inT;
  int lane = threadIdx.x & 31, wave = threadIdx.x >> 5;
  int m0 = blockIdx.x * 16;
  int n0 = blockIdx.y * 128 + wave * 16;      // 0..511 (H*E)
  v8f acc = {};
#pragma unroll
  for (int k0 = 0; k0 < H_; k0 += 32)
    acc = wmma_bf16(load_afrag(hbf, H_, m0, k0, lane),
                    load_bfrag(wt, H_, n0, k0, lane), acc);
  int col = n0 + (lane & 15);
  float bv = bias[col];
  int hh = col >> 2, e = col & 3;
#pragma unroll
  for (int r = 0; r < 8; ++r) {
    int row = m0 + r + ((lane >> 4) << 3);
    int bb = row >> 9, nn = row & (N_ - 1);
    dstT[((size_t)bb * H_ + hh) * NE_ + e * N_ + nn] = (__bf16)(acc[r] + bv);
  }
}

// ---- GEMM 2 (dominant): a_{in,out}[b] = m_{in,out}[b] (512x2048 f32) @ states
// Register-blocked: each wave computes a 16x64 tile (4 independent accs),
// reusing one fp32->bf16 A fragment across 4 B fragments.
// Block = 8 waves = 4 row-tiles x 2 col-groups -> 64x128 output per block.
__global__ void k_adj_gemm(const float* __restrict__ m,
                           const __bf16* __restrict__ inT,
                           const __bf16* __restrict__ outT,
                           float* __restrict__ a_in,
                           float* __restrict__ a_out) {
  int b = blockIdx.y, sel = blockIdx.z;
  int lane = threadIdx.x & 31, wave = threadIdx.x >> 5;
  int m0 = blockIdx.x * 64 + (wave & 3) * 16;   // row tile within batch
  int n0 = (wave >> 2) * 64;                    // col group (H = 2 x 64)
  const float*  A  = m + (size_t)b * N_ * (2 * NE_) + (sel ? NE_ : 0);
  const __bf16* Bt = (sel ? outT : inT) + (size_t)b * H_ * NE_;
  float*        C  = (sel ? a_out : a_in) + (size_t)b * N_ * H_;
  v8f acc0 = {}, acc1 = {}, acc2 = {}, acc3 = {};
  for (int k0 = 0; k0 < NE_; k0 += 32) {
    __builtin_prefetch(A + (size_t)(m0 + (lane & 15)) * (2 * NE_) + k0 + 96, 0, 0);
    v16bf af = load_afrag_f32(A, 2 * NE_, m0, k0, lane);
    acc0 = wmma_bf16(af, load_bfrag(Bt, NE_, n0,      k0, lane), acc0);
    acc1 = wmma_bf16(af, load_bfrag(Bt, NE_, n0 + 16, k0, lane), acc1);
    acc2 = wmma_bf16(af, load_bfrag(Bt, NE_, n0 + 32, k0, lane), acc2);
    acc3 = wmma_bf16(af, load_bfrag(Bt, NE_, n0 + 48, k0, lane), acc3);
  }
  int cb = n0 + (lane & 15);
#pragma unroll
  for (int r = 0; r < 8; ++r) {
    int row = m0 + r + ((lane >> 4) << 3);
    float* crow = C + (size_t)row * H_;
    crow[cb]      = acc0[r];
    crow[cb + 16] = acc1[r];
    crow[cb + 32] = acc2[r];
    crow[cb + 48] = acc3[r];
  }
}

// ---- cat = bf16([a_in | a_out | h]) ---------------------------------------
__global__ void k_build_cat(__bf16* __restrict__ cat, const float* __restrict__ a_in,
                            const float* __restrict__ a_out, const float* __restrict__ h) {
  int i = blockIdx.x * blockDim.x + threadIdx.x;
  if (i >= ROWS_ * K3H_) return;
  int g = i / K3H_, c = i - g * K3H_;
  float v = (c < H_)     ? a_in[(size_t)g * H_ + c]
          : (c < 2 * H_) ? a_out[(size_t)g * H_ + c - H_]
                         : h[(size_t)g * H_ + c - 2 * H_];
  cat[i] = (__bf16)v;
}

// ---- GEMM 3: z = sigmoid(cat@Wz+bz);  rh = bf16(sigmoid(cat@Wr+br) * h) ----
__global__ void k_zr_gemm(const __bf16* __restrict__ cat,
                          const __bf16* __restrict__ WzT, const __bf16* __restrict__ WrT,
                          const float* __restrict__ b_z, const float* __restrict__ b_r,
                          const float* __restrict__ h,
                          float* __restrict__ z, __bf16* __restrict__ rh) {
  int sel = blockIdx.z;
  const __bf16* wt   = sel ? WrT : WzT;
  const float*  bias = sel ? b_r : b_z;
  int lane = threadIdx.x & 31, wave = threadIdx.x >> 5;
  int m0 = blockIdx.x * 16, n0 = wave * 16;
  v8f acc = {};
#pragma unroll
  for (int k0 = 0; k0 < K3H_; k0 += 32)
    acc = wmma_bf16(load_afrag(cat, K3H_, m0, k0, lane),
                    load_bfrag(wt, K3H_, n0, k0, lane), acc);
  int col = n0 + (lane & 15);
  float bv = bias[col];
#pragma unroll
  for (int r = 0; r < 8; ++r) {
    int row = m0 + r + ((lane >> 4) << 3);
    size_t idx = (size_t)row * H_ + col;
    float s = sigmoidf_(acc[r] + bv);
    if (!sel) z[idx] = s;
    else      rh[idx] = (__bf16)(s * h[idx]);
  }
}

// ---- GEMM 4: h_hat = tanh([a_in|a_out|r*h]@Wt+bt); h = (1-z)h + z h_hat ----
__global__ void k_hhat_gemm(const __bf16* __restrict__ cat, const __bf16* __restrict__ rh,
                            const __bf16* __restrict__ WtT, const float* __restrict__ b_t,
                            const float* __restrict__ z, float* __restrict__ h) {
  int lane = threadIdx.x & 31, wave = threadIdx.x >> 5;
  int m0 = blockIdx.x * 16, n0 = wave * 16;
  v8f acc = {};
#pragma unroll
  for (int k0 = 0; k0 < K3H_; k0 += 32) {
    v16bf af = (k0 < 2 * H_) ? load_afrag(cat, K3H_, m0, k0, lane)
                             : load_afrag(rh, H_, m0, k0 - 2 * H_, lane);
    acc = wmma_bf16(af, load_bfrag(WtT, K3H_, n0, k0, lane), acc);
  }
  int col = n0 + (lane & 15);
  float bv = b_t[col];
#pragma unroll
  for (int r = 0; r < 8; ++r) {
    int row = m0 + r + ((lane >> 4) << 3);
    size_t idx = (size_t)row * H_ + col;
    float hhat = tanhf(acc[r] + bv);
    float zz = z[idx], hold = h[idx];
    h[idx] = hold + zz * (hhat - hold);
  }
}

// ---- readout ---------------------------------------------------------------
__global__ void k_build_catF(__bf16* __restrict__ catF, const float* __restrict__ h,
                             const float* __restrict__ a) {
  int i = blockIdx.x * blockDim.x + threadIdx.x;
  if (i >= ROWS_ * KF_) return;
  int g = i / KF_, c = i - g * KF_;
  float v = (c < H_) ? h[(size_t)g * H_ + c] : ((c == H_) ? a[g] : 0.0f);
  catF[i] = (__bf16)v;
}

__global__ void k_feat_gemm(const __bf16* __restrict__ catF, const __bf16* __restrict__ W1T,
                            const float* __restrict__ b1, float* __restrict__ feat) {
  int lane = threadIdx.x & 31, wave = threadIdx.x >> 5;
  int m0 = blockIdx.x * 16, n0 = wave * 16;
  v8f acc = {};
#pragma unroll
  for (int k0 = 0; k0 < KF_; k0 += 32)
    acc = wmma_bf16(load_afrag(catF, KF_, m0, k0, lane),
                    load_bfrag(W1T, KF_, n0, k0, lane), acc);
  int col = n0 + (lane & 15);
  float bv = b1[col];
#pragma unroll
  for (int r = 0; r < 8; ++r) {
    int row = m0 + r + ((lane >> 4) << 3);
    feat[(size_t)row * H_ + col] = tanhf(acc[r] + bv);
  }
}

__global__ void k_out(const float* __restrict__ feat, const float* __restrict__ W2,
                      const float* __restrict__ b2, float* __restrict__ out) {
  int lane = threadIdx.x & 31;
  int g = blockIdx.x * 8 + (threadIdx.x >> 5);
  float v = 0.f;
#pragma unroll
  for (int c = 0; c < H_; c += 32) v += feat[(size_t)g * H_ + c + lane] * W2[c + lane];
#pragma unroll
  for (int off = 16; off > 0; off >>= 1) v += __shfl_xor(v, off, 32);
  if (lane == 0) out[g] = v + b2[0];
}

// ---------------------------------------------------------------------------
extern "C" void kernel_launch(void* const* d_in, const int* in_sizes, int n_in,
                              void* d_out, int out_size, void* d_ws, size_t ws_size,
                              hipStream_t stream) {
  (void)in_sizes; (void)n_in; (void)out_size; (void)ws_size;
  const float* x    = (const float*)d_in[0];
  const float* a    = (const float*)d_in[1];
  const float* m    = (const float*)d_in[2];
  const float* W_in = (const float*)d_in[3];
  const float* b_in = (const float*)d_in[4];
  const float* W_out= (const float*)d_in[5];
  const float* b_out= (const float*)d_in[6];
  const float* W_z  = (const float*)d_in[7];
  const float* b_z  = (const float*)d_in[8];
  const float* W_r  = (const float*)d_in[9];
  const float* b_r  = (const float*)d_in[10];
  const float* W_t  = (const float*)d_in[11];
  const float* b_t  = (const float*)d_in[12];
  const float* W1   = (const float*)d_in[13];
  const float* b1   = (const float*)d_in[14];
  const float* W2   = (const float*)d_in[15];
  const float* b2   = (const float*)d_in[16];

  size_t off = 0;
  auto carve = [&](size_t bytes) -> char* {
    char* p = (char*)d_ws + off;
    off += (bytes + 255) & ~(size_t)255;
    return p;
  };
  float*  h    = (float*)carve((size_t)ROWS_ * H_ * 4);
  __bf16* hbf  = (__bf16*)carve((size_t)ROWS_ * H_ * 2);
  __bf16* WinT = (__bf16*)carve((size_t)512 * 128 * 2);
  __bf16* WoutT= (__bf16*)carve((size_t)512 * 128 * 2);
  __bf16* WzT  = (__bf16*)carve((size_t)128 * K3H_ * 2);
  __bf16* WrT  = (__bf16*)carve((size_t)128 * K3H_ * 2);
  __bf16* WtT  = (__bf16*)carve((size_t)128 * K3H_ * 2);
  __bf16* W1T  = (__bf16*)carve((size_t)128 * KF_ * 2);
  __bf16* inT  = (__bf16*)carve((size_t)B_ * H_ * NE_ * 2);
  __bf16* outT = (__bf16*)carve((size_t)B_ * H_ * NE_ * 2);
  float*  ain  = (float*)carve((size_t)ROWS_ * H_ * 4);
  float*  aout = (float*)carve((size_t)ROWS_ * H_ * 4);
  float*  z    = (float*)carve((size_t)ROWS_ * H_ * 4);
  __bf16* rh   = (__bf16*)carve((size_t)ROWS_ * H_ * 2);
  __bf16* cat  = (__bf16*)carve((size_t)ROWS_ * K3H_ * 2);
  __bf16* catF = (__bf16*)carve((size_t)ROWS_ * KF_ * 2);
  float*  feat = (float*)carve((size_t)ROWS_ * H_ * 4);

  const int T = 256;
  // prep: h = x, bf16-transposed weights
  k_copy_f32<<<(ROWS_ * H_ + T - 1) / T, T, 0, stream>>>(h, x, ROWS_ * H_);
  k_transpose_bf<<<(512 * 128 + T - 1) / T, T, 0, stream>>>(WinT,  W_in,  128, 512, 128);
  k_transpose_bf<<<(512 * 128 + T - 1) / T, T, 0, stream>>>(WoutT, W_out, 128, 512, 128);
  k_transpose_bf<<<(128 * K3H_ + T - 1) / T, T, 0, stream>>>(WzT, W_z, K3H_, 128, K3H_);
  k_transpose_bf<<<(128 * K3H_ + T - 1) / T, T, 0, stream>>>(WrT, W_r, K3H_, 128, K3H_);
  k_transpose_bf<<<(128 * K3H_ + T - 1) / T, T, 0, stream>>>(WtT, W_t, K3H_, 128, K3H_);
  k_transpose_bf<<<(128 * KF_ + T - 1) / T, T, 0, stream>>>(W1T, W1, H_ + 1, 128, KF_);

  for (int step = 0; step < STEPS_; ++step) {
    k_f32_to_bf<<<(ROWS_ * H_ + T - 1) / T, T, 0, stream>>>(hbf, h, ROWS_ * H_);
    k_states_gemm<<<dim3(ROWS_ / 16, 4, 2), T, 0, stream>>>(hbf, WinT, WoutT, b_in, b_out, inT, outT);
    k_adj_gemm<<<dim3(N_ / 64, B_, 2), T, 0, stream>>>(m, inT, outT, ain, aout);
    k_build_cat<<<(ROWS_ * K3H_ + T - 1) / T, T, 0, stream>>>(cat, ain, aout, h);
    k_zr_gemm<<<dim3(ROWS_ / 16, 1, 2), T, 0, stream>>>(cat, WzT, WrT, b_z, b_r, h, z, rh);
    k_hhat_gemm<<<dim3(ROWS_ / 16), T, 0, stream>>>(cat, rh, WtT, b_t, z, h);
  }
  k_build_catF<<<(ROWS_ * KF_ + T - 1) / T, T, 0, stream>>>(catF, h, a);
  k_feat_gemm<<<dim3(ROWS_ / 16), T, 0, stream>>>(catF, W1T, b1, feat);
  k_out<<<ROWS_ / 8, T, 0, stream>>>(feat, W2, b2, (float*)d_out);
}
